// DeFlowNet_11570641896076
// MI455X (gfx1250) — matrix operand order repaired and checked by reading, not compiled, not run on hardware
//
#include <hip/hip_runtime.h>
#include <hip/hip_bf16.h>
#include <math.h>

// =====================================================================
// DeFlowNet (PWC-style flow+depth) forward for MI455X / gfx1250.
//  - Convs = implicit GEMM on v_wmma_f32_16x16x32_f16 (f32 accum).
//  - 64x64 macro tile per 128-thread block: 4 waves x 4 N-tiles = 16 WMMA
//    per 32-wide K chunk.
//  - Weights pre-converted to f16; B tiles DMA'd into LDS with the CDNA5
//    Tensor Data Mover (tensor_load_to_lds + s_wait_tensorcnt).
//  - Double-buffered LDS tiles: chunk k+1 staging (A im2col gather +
//    TDM B transfer) overlaps the chunk-k WMMA chain.
// =====================================================================

typedef __attribute__((ext_vector_type(16))) _Float16 v16h;
typedef __attribute__((ext_vector_type(8)))  _Float16 v8h;
typedef __attribute__((ext_vector_type(8)))  float    v8f;
typedef __attribute__((ext_vector_type(4)))  unsigned int uint32x4;
typedef __attribute__((ext_vector_type(4)))  int      int32x4;
typedef __attribute__((ext_vector_type(8)))  int      int32x8;

#define ACT_NONE    0
#define ACT_LEAKY   1
#define ACT_SIGMOID 2

// ---------------------------------------------------------------
// TDM: load a [64 rows x 32 cols] f16 tile (row stride = strideHalves)
// from global memory into LDS at byte offset lds_off. Rows/cols beyond
// (rowsRem, colsRem) are zero-filled by tensor-dim OOB clamping.
// ---------------------------------------------------------------
__device__ __forceinline__ void tdm_load_b_tile(unsigned lds_off,
                                                const _Float16* gsrc,
                                                int rowsRem, int colsRem,
                                                int strideHalves) {
  unsigned long long ga = (unsigned long long)(uintptr_t)gsrc;
  uint32x4 g0;
  g0[0] = 1u;                                  // count=1 (valid user D#)
  g0[1] = lds_off;                             // lds_addr (bytes)
  g0[2] = (unsigned)(ga & 0xffffffffull);      // global_addr[31:0]
  g0[3] = (unsigned)((ga >> 32) & 0x1ffffffull) | (2u << 30); // addr[56:32], type=2
  unsigned td0 = (unsigned)colsRem;            // tensor_dim0 (halves)
  unsigned td1 = (unsigned)rowsRem;            // tensor_dim1 (rows)
  int32x8 g1;
  g1[0] = (int)(1u << 16);                     // data_size=1 (2 bytes/elem)
  g1[1] = (int)(td0 << 16);                    // tensor_dim0[15:0] @ bits 63:48
  g1[2] = (int)((td0 >> 16) | (td1 << 16));    // td0 hi, td1 lo
  g1[3] = (int)(((td1 >> 16) & 0xffffu) | (32u << 16)); // td1 hi, tile_dim0=32
  g1[4] = 64;                                  // tile_dim1=64, tile_dim2=0
  g1[5] = (int)(unsigned)strideHalves;         // tensor_dim0_stride[31:0]
  g1[6] = 0;
  g1[7] = 0;
  int32x4 z4 = {0, 0, 0, 0};
#if !defined(__clang_major__) || __clang_major__ <= 22
  __builtin_amdgcn_tensor_load_to_lds(g0, g1, z4, z4, 0);
#else
  int32x8 z8 = {0, 0, 0, 0, 0, 0, 0, 0};
  __builtin_amdgcn_tensor_load_to_lds(g0, g1, z4, z4, z8, 0);
#endif
}

// strength-reduced im2col gather of one 64x32 f16 A tile chunk into LDS;
// this thread owns K-slot (k0+lane) and rows m = wave + 4j.
__device__ __forceinline__ void stage_a_chunk(
    const float* __restrict__ in, _Float16* __restrict__ At,
    int k0, int K, int M, int tileM, int wave, int lane,
    int oy0, int ox0, int Hin, int Win, int Wout,
    int stride, int dil, int pad)
{
  const int kA = k0 + lane;
  int ci = 0, iyOff = 0, ixOff = 0;
  bool kvalid = kA < K;
  if (kvalid) {
    ci = kA / 9;
    int r = kA - ci * 9;
    int ky = r / 3, kx = r - ky * 3;
    iyOff = ky * dil - pad;
    ixOff = kx * dil - pad;
  }
  const float* inC = in + (size_t)ci * Hin * Win;
  int oy = oy0, ox = ox0;
  #pragma unroll 4
  for (int j = 0; j < 16; ++j) {
    int m = wave + 4 * j;
    float v = 0.0f;
    if (kvalid && (tileM + m) < M) {
      int iy = oy * stride + iyOff;
      int ix = ox * stride + ixOff;
      if (iy >= 0 && iy < Hin && ix >= 0 && ix < Win)
        v = inC[iy * Win + ix];
    }
    At[m * 32 + lane] = (_Float16)v;
    ox += 4;
    if (ox >= Wout) { ox -= Wout; ++oy; }
  }
}

__device__ __forceinline__ v16h combine16(v8h lo, v8h hi) {
  v16h r;
  #pragma unroll
  for (int e = 0; e < 8; ++e) { r[e] = lo[e]; r[8 + e] = hi[e]; }
  return r;
}

// ------------------------------------------------------------------
// Implicit-GEMM 3x3 conv (stride/dilation), NCHW f32 in / f32 out,
// f16 WMMA with f32 accumulate.  M = Hout*Wout, N = Cout, K = Cin*9.
// Block = 128 threads = 4 waves. Block tile: M=64 (wave w -> rows
// w*16..w*16+15), N=64 (4 accumulators / 4 wmma per wave per chunk).
// Double-buffered LDS; staging of chunk c+1 overlaps WMMA of chunk c.
// ------------------------------------------------------------------
__global__ __launch_bounds__(128) void k_conv3x3_wmma(
    const float* __restrict__ in, const _Float16* __restrict__ wgt,
    const float* __restrict__ bias, float* __restrict__ out,
    int Cin, int Cout, int Hin, int Win, int Hout, int Wout,
    int stride, int dil, int act)
{
  const int K      = Cin * 9;
  const int M      = Hout * Wout;
  const int lane   = threadIdx.x & 31;
  const int wave   = threadIdx.x >> 5;
  const int tileM  = blockIdx.x * 64;
  const int nBase0 = blockIdx.y * 64;
  const int pad    = dil;

  __shared__ __align__(16) _Float16 Atile[2][64 * 32]; // double-buffered
  __shared__ __align__(16) _Float16 Btile[2][64 * 32];

  const unsigned bLds0 = (unsigned)(uintptr_t)(&Btile[0][0]);
  const unsigned bLds1 = (unsigned)(uintptr_t)(&Btile[1][0]);

  v8f acc[4] = {{}, {}, {}, {}};
  const int half = lane >> 4;        // K-half selector
  const int mrow = wave * 16 + (lane & 15);

  // spatial decode for this thread's staged rows (m = wave + 4j), hoisted
  const int gm0 = tileM + wave;
  const int oy0 = gm0 / Wout;
  const int ox0 = gm0 - oy0 * Wout;

  const int nCh = (K + 31) / 32;
  const _Float16* wRow = wgt + (size_t)nBase0 * K;

  // ---- prologue: stage chunk 0 into buffer 0 ----
  if (wave == 0)
    tdm_load_b_tile(bLds0, wRow, Cout - nBase0, K, K);
  stage_a_chunk(in, &Atile[0][0], 0, K, M, tileM, wave, lane,
                oy0, ox0, Hin, Win, Wout, stride, dil, pad);
  if (wave == 0) __builtin_amdgcn_s_wait_tensorcnt(0);
  __syncthreads();

  for (int c = 0; c < nCh; ++c) {
    const int buf = c & 1;

    // ---- fragments for chunk c (all loads issued up front) ----
    const v8h* arow = (const v8h*)(&Atile[buf][mrow * 32]);
    v16h a = combine16(arow[half], arow[2 + half]);
    v16h b[4];
    #pragma unroll
    for (int t = 0; t < 4; ++t) {
      const v8h* brow =
          (const v8h*)(&Btile[buf][(t * 16 + (lane & 15)) * 32 + half * 8]);
      b[t] = combine16(brow[0], brow[2]);
    }

    // ---- stage chunk c+1 into the other buffer (overlaps WMMA) ----
    const bool hasNext = (c + 1) < nCh;
    if (hasNext) {
      const int k0n = (c + 1) * 32;
      if (wave == 0)
        tdm_load_b_tile(buf ? bLds0 : bLds1, wRow + k0n,
                        Cout - nBase0, K - k0n, K);
      stage_a_chunk(in, &Atile[buf ^ 1][0], k0n, K, M, tileM, wave, lane,
                    oy0, ox0, Hin, Win, Wout, stride, dil, pad);
    }

    // ---- 4 back-to-back WMMAs ----
    #pragma unroll
    for (int t = 0; t < 4; ++t)
      acc[t] = __builtin_amdgcn_wmma_f32_16x16x32_f16(
          false, a, false, b[t], (short)0, acc[t], false, false);

    if (wave == 0 && hasNext) __builtin_amdgcn_s_wait_tensorcnt(0);
    __syncthreads();
  }

  // ---- epilogue: bias + activation + stores ----
  #pragma unroll
  for (int t = 0; t < 4; ++t) {
    int ncol = nBase0 + t * 16 + (lane & 15);
    if (ncol < Cout) {
      float bv = bias[ncol];
      #pragma unroll
      for (int r = 0; r < 8; ++r) {
        int gm = tileM + wave * 16 + r + 8 * half;
        if (gm < M) {
          float v = acc[t][r] + bv;
          if (act == ACT_LEAKY)        v = v > 0.0f ? v : 0.1f * v;
          else if (act == ACT_SIGMOID) v = 1.0f / (1.0f + expf(-v));
          out[(size_t)ncol * M + gm] = v;
        }
      }
    }
  }
}

// ------------------------- elementwise helpers -------------------------
__global__ void k_cvt_f16(const float* __restrict__ a, _Float16* __restrict__ b, int n) {
  int i = blockIdx.x * blockDim.x + threadIdx.x;
  if (i < n) b[i] = (_Float16)a[i];
}
__global__ void k_copy(const float* __restrict__ a, float* __restrict__ b, int n) {
  int i = blockIdx.x * blockDim.x + threadIdx.x;
  if (i < n) b[i] = a[i];
}
__global__ void k_fill0(float* p, int n) {
  int i = blockIdx.x * blockDim.x + threadIdx.x;
  if (i < n) p[i] = 0.0f;
}
__global__ void k_add_inplace(float* __restrict__ a, const float* __restrict__ b, int n) {
  int i = blockIdx.x * blockDim.x + threadIdx.x;
  if (i < n) a[i] += b[i];
}
__global__ void k_sigmoid(float* a, int n) {
  int i = blockIdx.x * blockDim.x + threadIdx.x;
  if (i < n) a[i] = 1.0f / (1.0f + expf(-a[i]));
}
__global__ void k_normalize(float* x, const float* __restrict__ ms, int n) {
  int i = blockIdx.x * blockDim.x + threadIdx.x;
  if (i < n) x[i] = (x[i] - ms[0]) / ms[1];
}

// nearest 2x upsample (jnp.repeat x2 on H and W)
__global__ void k_upsample2x(const float* __restrict__ in, float* __restrict__ out,
                             int C, int H, int W) {
  int i = blockIdx.x * blockDim.x + threadIdx.x;
  int W2 = 2 * W, H2 = 2 * H;
  int n = C * H2 * W2;
  if (i >= n) return;
  int x = i % W2;
  int t = i / W2;
  int y = t % H2;
  int c = t / H2;
  out[i] = in[(c * H + (y >> 1)) * W + (x >> 1)];
}

// bilinear resize matching reference resize_as()
__global__ void k_resize_bilinear(const float* __restrict__ in, float* __restrict__ out,
                                  int C, int h, int w, int H, int W,
                                  float sy, float sx) {
  int i = blockIdx.x * blockDim.x + threadIdx.x;
  int n = C * H * W;
  if (i >= n) return;
  int x = i % W;
  int t = i / W;
  int y = t % H;
  int c = t / H;
  float iy = y * sy, ix = x * sx;
  int y0 = (int)floorf(iy); if (y0 < 0) y0 = 0; if (y0 > h - 1) y0 = h - 1;
  int x0 = (int)floorf(ix); if (x0 < 0) x0 = 0; if (x0 > w - 1) x0 = w - 1;
  int y1 = y0 + 1; if (y1 > h - 1) y1 = h - 1;
  int x1 = x0 + 1; if (x1 > w - 1) x1 = w - 1;
  float wy = iy - (float)y0, wx = ix - (float)x0;
  const float* p = in + c * h * w;
  float a = p[y0 * w + x0], b = p[y0 * w + x1];
  float cc = p[y1 * w + x0], d = p[y1 * w + x1];
  out[i] = a * (1 - wy) * (1 - wx) + b * (1 - wy) * wx +
           cc * wy * (1 - wx) + d * wy * wx;
}

// bilinear warp with zero-outside; optional validity mask (m>=1) multiply
__global__ void k_warp(const float* __restrict__ x, const float* __restrict__ flow,
                       float* __restrict__ out, int C, int H, int W, int useMask) {
  int i = blockIdx.x * blockDim.x + threadIdx.x;
  int n = C * H * W;
  if (i >= n) return;
  int px = i % W;
  int t  = i / W;
  int py = t % H;
  int c  = t / H;
  int HW = H * W;
  int idx = py * W + px;
  float ix = (float)px + flow[idx];
  float iy = (float)py + flow[HW + idx];
  float x0f = floorf(ix), y0f = floorf(iy);
  float wx = ix - x0f, wy = iy - y0f;
  int x0 = (int)x0f, y0 = (int)y0f;
  float s = 0.0f, m = 0.0f;
  const float* p = x + c * HW;
  #pragma unroll
  for (int dy = 0; dy < 2; ++dy) {
    #pragma unroll
    for (int dx = 0; dx < 2; ++dx) {
      int xx = x0 + dx, yy = y0 + dy;
      float wgt = (dx ? wx : 1.0f - wx) * (dy ? wy : 1.0f - wy);
      if (xx >= 0 && xx < W && yy >= 0 && yy < H) {
        s += wgt * p[yy * W + xx];
        m += wgt;
      }
    }
  }
  if (useMask) s = (m >= 1.0f - 1e-5f) ? s : 0.0f;
  out[i] = s;
}

// 9x9 cost volume (mean over C) + leaky relu 0.1 fused
__global__ void k_cost_volume(const float* __restrict__ f1, const float* __restrict__ f2,
                              float* __restrict__ out, int C, int H, int W) {
  int i = blockIdx.x * blockDim.x + threadIdx.x;
  int n = 81 * H * W;
  if (i >= n) return;
  int x = i % W;
  int t = i / W;
  int y = t % H;
  int q = t / H;
  int sy = q / 9 - 4, sx = q % 9 - 4;
  int yy = y + sy, xx = x + sx;
  float s = 0.0f;
  if (yy >= 0 && yy < H && xx >= 0 && xx < W) {
    int HW = H * W;
    int i1 = y * W + x, i2 = yy * W + xx;
    for (int c = 0; c < C; ++c) s += f1[c * HW + i1] * f2[c * HW + i2];
    s /= (float)C;
  }
  out[i] = s > 0.0f ? s : 0.1f * s;
}

// sum / sumsq reduction (LDS tree + global f32 atomics)
__global__ void k_reduce_stats(const float* __restrict__ x, int n, float* __restrict__ slot) {
  __shared__ float ssum[256];
  __shared__ float ssq[256];
  float s = 0.0f, q = 0.0f;
  for (int i = blockIdx.x * blockDim.x + threadIdx.x; i < n;
       i += gridDim.x * blockDim.x) {
    float v = x[i];
    s += v; q += v * v;
  }
  ssum[threadIdx.x] = s; ssq[threadIdx.x] = q;
  __syncthreads();
  for (int off = 128; off > 0; off >>= 1) {
    if (threadIdx.x < (unsigned)off) {
      ssum[threadIdx.x] += ssum[threadIdx.x + off];
      ssq[threadIdx.x]  += ssq[threadIdx.x + off];
    }
    __syncthreads();
  }
  if (threadIdx.x == 0) {
    atomicAdd(&slot[0], ssum[0]);
    atomicAdd(&slot[1], ssq[0]);
  }
}

// m = mean of 4 means, s = mean of 4 stds (ddof=1, +1e-16)
__global__ void k_combine_stats(const float* __restrict__ slots, float* __restrict__ ms,
                                float invN, float invNm1) {
  if (blockIdx.x == 0 && threadIdx.x == 0) {
    float m = 0.0f, s = 0.0f;
    for (int t = 0; t < 4; ++t) {
      float sum = slots[2 * t], sq = slots[2 * t + 1];
      float mean = sum * invN;
      float var = (sq - sum * mean) * invNm1;
      m += mean;
      s += sqrtf(var + 1e-16f);
    }
    ms[0] = m * 0.25f;
    ms[1] = s * 0.25f;
  }
}

// final static/occlusion masks; d_out layout: ff(2HW) fb(2HW) d1 d2 static fwd bwd
__global__ void k_final(float* __restrict__ o, const float* __restrict__ wb,
                        const float* __restrict__ wf, int HW) {
  int i = blockIdx.x * blockDim.x + threadIdx.x;
  if (i >= HW) return;
  float ffx = o[i],          ffy = o[HW + i];
  float fbx = o[2 * HW + i], fby = o[3 * HW + i];
  float mf = sqrtf(ffx * ffx + ffy * ffy);
  float mb = sqrtf(fbx * fbx + fby * fby);
  o[6 * HW + i] = (mb + mf <= 2.0f) ? 1.0f : 0.0f;
  float thr = 0.01f * (mf + mb) + 0.5f;
  float ax = ffx + wb[i], ay = ffy + wb[HW + i];
  o[7 * HW + i] = (sqrtf(ax * ax + ay * ay) > thr) ? 1.0f : 0.0f;
  float bx = fbx + wf[i], by = fby + wf[HW + i];
  o[8 * HW + i] = (sqrtf(bx * bx + by * by) > thr) ? 1.0f : 0.0f;
}

// =====================================================================
// Host-side orchestration
// =====================================================================
struct Conv { const float* w; const float* b; _Float16* wh; int cin, cout; };
struct Pyr  { float* lv[5]; };
struct Carry { float *ff, *fb, *d1, *d2, *xo1, *xo2; };

static void convLaunch(hipStream_t st, const float* in, const Conv& c, float* out,
                       int Hin, int Win, int stride, int dil, int act) {
  int Hout = (Hin - 1) / stride + 1;
  int Wout = (Win - 1) / stride + 1;
  int M = Hout * Wout;
  dim3 blk(128);
  dim3 grd((M + 63) / 64, (c.cout + 63) / 64);
  k_conv3x3_wmma<<<grd, blk, 0, st>>>(in, c.wh, c.b, out, c.cin, c.cout,
                                      Hin, Win, Hout, Wout, stride, dil, act);
}

#define EW(kern, n, ...) kern<<<dim3(((n) + 255) / 256), dim3(256), 0, stream>>>(__VA_ARGS__)

extern "C" void kernel_launch(void* const* d_in, const int* in_sizes, int n_in,
                              void* d_out, int out_size, void* d_ws, size_t ws_size,
                              hipStream_t stream) {
  (void)in_sizes; (void)n_in; (void)out_size; (void)ws_size;

  // ---------------- workspace bump allocator ----------------
  char* base = (char*)d_ws;
  size_t off = 0;
  auto alloc = [&](size_t nf) -> float* {
    float* p = (float*)(base + off);
    off += ((nf * sizeof(float) + 255) & ~(size_t)255);
    return p;
  };

  // ---------------- parse params (jax sorted-key tree flatten) ----------------
  // order: ctx{convs,depth,flow}, dec[5]{convs,depth,flow},
  //        depth_feat[5]{c1,c2}, feat[5]{c1,c2}, upconv[4]; each conv: b then w
  int cur = 4;
  auto nc = [&](int ci, int co) {
    Conv c;
    c.b = (const float*)d_in[cur++];
    c.w = (const float*)d_in[cur++];
    c.cin = ci; c.cout = co;
    // pre-convert weights to f16 [Cout][K] for the WMMA/TDM path
    int n = co * ci * 9;
    c.wh = (_Float16*)alloc((size_t)(n + 1) / 2);
    EW(k_cvt_f16, n, c.w, c.wh, n);
    return c;
  };
  Conv ctxC[6], ctxDepth, ctxFlow;
  Conv decC[5][3], decDepth[5], decFlow[5];
  Conv dfeat[10], feat[10], upc[4];
  {
    const int cci[6] = {99, 128, 128, 128, 96, 64};
    const int cco[6] = {128, 128, 128, 96, 64, 32};
    for (int i = 0; i < 6; ++i) ctxC[i] = nc(cci[i], cco[i]);
    ctxDepth = nc(32, 1);
    ctxFlow  = nc(32, 2);
    const int chin[5] = {465, 436, 372, 308, 244};
    for (int l = 0; l < 5; ++l) {
      decC[l][0] = nc(chin[l], 128);
      decC[l][1] = nc(128, 128);
      decC[l][2] = nc(128, 96);
      decDepth[l] = nc(96, 1);
      decFlow[l]  = nc(96, 2);
    }
    const int dch[6] = {1, 32, 64, 96, 128, 192};
    for (int b = 0; b < 5; ++b) {
      dfeat[2 * b]     = nc(dch[b], dch[b + 1]);
      dfeat[2 * b + 1] = nc(dch[b + 1], dch[b + 1]);
    }
    const int ich[6] = {3, 32, 64, 96, 128, 192};
    for (int b = 0; b < 5; ++b) {
      feat[2 * b]     = nc(ich[b], ich[b + 1]);
      feat[2 * b + 1] = nc(ich[b + 1], ich[b + 1]);
    }
    for (int i = 0; i < 4; ++i) upc[i] = nc(96, 96);
  }

  const int MAXHW = 96 * 320;
  // persistent carry sets (ping-pong across levels)
  Carry setA, setB;
  for (Carry* s : {&setA, &setB}) {
    s->ff = alloc(2 * MAXHW); s->fb = alloc(2 * MAXHW);
    s->d1 = alloc(MAXHW);     s->d2 = alloc(MAXHW);
    s->xo1 = alloc(96 * MAXHW); s->xo2 = alloc(96 * MAXHW);
  }
  float* ctmp = alloc(32 * 96 * 320); // reusable c1 output buffer

  // ---------------- feature pyramids ----------------
  Pyr px1, px2, pd1, pd2;
  auto extract = [&](const float* inp, Conv* blocks, Pyr& pyr) {
    const float* x = inp;
    int H = 192, W = 640;
    for (int b = 0; b < 5; ++b) {
      int Ho = (H - 1) / 2 + 1, Wo = (W - 1) / 2 + 1;
      convLaunch(stream, x, blocks[2 * b], ctmp, H, W, 2, 1, ACT_LEAKY);
      pyr.lv[b] = alloc((size_t)blocks[2 * b + 1].cout * Ho * Wo);
      convLaunch(stream, ctmp, blocks[2 * b + 1], pyr.lv[b], Ho, Wo, 1, 1, ACT_LEAKY);
      x = pyr.lv[b]; H = Ho; W = Wo;
    }
  };
  extract((const float*)d_in[0], feat,  px1);
  extract((const float*)d_in[1], feat,  px2);
  extract((const float*)d_in[2], dfeat, pd1);
  extract((const float*)d_in[3], dfeat, pd2);

  // ---------------- coarse-to-fine level loop ----------------
  const int Hs[5] = {6, 12, 24, 48, 96};
  const int Ws[5] = {20, 40, 80, 160, 320};
  const int CcA[5] = {192, 128, 96, 64, 32};  // image feature channels
  const int CdA[5] = {192, 128, 96, 64, 32};  // depth feature channels
  Carry* P = &setA;
  Carry* Cur = &setB;

  for (int l = 0; l < 5; ++l) {
    const int H = Hs[l], W = Ws[l], HW = H * W;
    const int Cc = CcA[l], Cd = CdA[l];
    float* x1 = px1.lv[4 - l];
    float* x2 = px2.lv[4 - l];
    float* df1 = pd1.lv[4 - l];
    float* df2 = pd2.lv[4 - l];

    size_t levelMark = off;
    float* x1w = alloc((size_t)Cc * HW);
    float* x2w = alloc((size_t)Cc * HW);
    float* uc1 = nullptr;
    float* uc2 = nullptr;

    if (l == 0) {
      EW(k_copy, Cc * HW, x1, x1w, Cc * HW);
      EW(k_copy, Cc * HW, x2, x2w, Cc * HW);
    } else {
      int ph = Hs[l - 1], pw = Ws[l - 1];
      float sy = (float)(ph - 1) / (float)(H - 1);
      float sx = (float)(pw - 1) / (float)(W - 1);
      EW(k_resize_bilinear, 2 * HW, P->ff, Cur->ff, 2, ph, pw, H, W, sy, sx);
      EW(k_resize_bilinear, 2 * HW, P->fb, Cur->fb, 2, ph, pw, H, W, sy, sx);
      EW(k_resize_bilinear, HW,     P->d1, Cur->d1, 1, ph, pw, H, W, sy, sx);
      EW(k_resize_bilinear, HW,     P->d2, Cur->d2, 1, ph, pw, H, W, sy, sx);
      float* ut = alloc((size_t)96 * HW);
      uc1 = alloc((size_t)96 * HW);
      uc2 = alloc((size_t)96 * HW);
      EW(k_upsample2x, 96 * HW, P->xo1, ut, 96, ph, pw);
      convLaunch(stream, ut, upc[l - 1], uc1, H, W, 1, 1, ACT_LEAKY);
      EW(k_upsample2x, 96 * HW, P->xo2, ut, 96, ph, pw);
      convLaunch(stream, ut, upc[l - 1], uc2, H, W, 1, 1, ACT_LEAKY);
      EW(k_warp, Cc * HW, x2, Cur->ff, x2w, Cc, H, W, 1);
      EW(k_warp, Cc * HW, x1, Cur->fb, x1w, Cc, H, W, 1);
    }

    // joint feature normalization of {x1, x2, x1w, x2w}
    float* slots = alloc(8);
    float* ms = alloc(2);
    EW(k_fill0, 8, slots, 8);
    {
      int n = Cc * HW;
      int nb = (n + 255) / 256; if (nb > 1024) nb = 1024;
      k_reduce_stats<<<dim3(nb), dim3(256), 0, stream>>>(x1,  n, slots + 0);
      k_reduce_stats<<<dim3(nb), dim3(256), 0, stream>>>(x2,  n, slots + 2);
      k_reduce_stats<<<dim3(nb), dim3(256), 0, stream>>>(x1w, n, slots + 4);
      k_reduce_stats<<<dim3(nb), dim3(256), 0, stream>>>(x2w, n, slots + 6);
      k_combine_stats<<<1, 1, 0, stream>>>(slots, ms, 1.0f / (float)n,
                                           1.0f / (float)(n - 1));
      EW(k_normalize, n, x1,  ms, n);
      EW(k_normalize, n, x2,  ms, n);
      EW(k_normalize, n, x1w, ms, n);
      EW(k_normalize, n, x2w, ms, n);
    }

    float* corrF = alloc((size_t)81 * HW);
    float* corrB = alloc((size_t)81 * HW);
    EW(k_cost_volume, 81 * HW, x1, x2w, corrF, Cc, H, W);
    EW(k_cost_volume, 81 * HW, x2, x1w, corrB, Cc, H, W);

    const int chin = (l == 0) ? (81 + Cc + Cd) : (81 + Cc + 96 + 2 + 1 + Cd);

    for (int s = 0; s < 2; ++s) {
      float* corr   = s ? corrB : corrF;
      float* xf     = s ? x2 : x1;
      float* dd     = s ? df2 : df1;
      float* uc     = s ? uc2 : uc1;
      float* flowP  = s ? Cur->fb : Cur->ff;
      float* depthP = s ? Cur->d2 : Cur->d1;
      float* xoDst  = s ? Cur->xo2 : Cur->xo1;

      size_t streamMark = off;
      float* cat = alloc((size_t)chin * HW);
      int o = 0;
      EW(k_copy, 81 * HW, corr, cat + (size_t)o * HW, 81 * HW); o += 81;
      EW(k_copy, Cc * HW, xf,   cat + (size_t)o * HW, Cc * HW); o += Cc;
      if (l > 0) {
        EW(k_copy, 96 * HW, uc,     cat + (size_t)o * HW, 96 * HW); o += 96;
        EW(k_copy, 2 * HW,  flowP,  cat + (size_t)o * HW, 2 * HW);  o += 2;
        EW(k_copy, HW,      depthP, cat + (size_t)o * HW, HW);      o += 1;
      }
      EW(k_copy, Cd * HW, dd, cat + (size_t)o * HW, Cd * HW); o += Cd;

      float* t1 = alloc((size_t)128 * HW);
      float* t2 = alloc((size_t)128 * HW);
      convLaunch(stream, cat, decC[l][0], t1, H, W, 1, 1, ACT_LEAKY);
      convLaunch(stream, t1,  decC[l][1], t2, H, W, 1, 1, ACT_LEAKY);
      convLaunch(stream, t2,  decC[l][2], xoDst, H, W, 1, 1, ACT_LEAKY);
      float* fl = alloc((size_t)2 * HW);
      float* de = alloc(HW);
      convLaunch(stream, xoDst, decFlow[l],  fl, H, W, 1, 1, ACT_NONE);
      convLaunch(stream, xoDst, decDepth[l], de, H, W, 1, 1, ACT_NONE);

      if (l == 0) EW(k_copy, 2 * HW, fl, flowP, 2 * HW);
      else        EW(k_add_inplace, 2 * HW, flowP, fl, 2 * HW);
      EW(k_copy, HW, de, depthP, HW);

      if (l != 4) {
        EW(k_sigmoid, HW, depthP, HW);
      } else {
        // context network (dilated convs) on [xo(96), flow(2), depth(1)]
        float* cat99 = alloc((size_t)99 * HW);
        EW(k_copy, 96 * HW, xoDst,  cat99, 96 * HW);
        EW(k_copy, 2 * HW,  flowP,  cat99 + (size_t)96 * HW, 2 * HW);
        EW(k_copy, HW,      depthP, cat99 + (size_t)98 * HW, HW);
        convLaunch(stream, cat99, ctxC[0], t1, H, W, 1, 1,  ACT_LEAKY);
        convLaunch(stream, t1,    ctxC[1], t2, H, W, 1, 2,  ACT_LEAKY);
        convLaunch(stream, t2,    ctxC[2], t1, H, W, 1, 4,  ACT_LEAKY);
        convLaunch(stream, t1,    ctxC[3], t2, H, W, 1, 8,  ACT_LEAKY);
        convLaunch(stream, t2,    ctxC[4], t1, H, W, 1, 16, ACT_LEAKY);
        convLaunch(stream, t1,    ctxC[5], t2, H, W, 1, 1,  ACT_LEAKY);
        float* fres = alloc((size_t)2 * HW);
        convLaunch(stream, t2, ctxFlow, fres, H, W, 1, 1, ACT_NONE);
        EW(k_add_inplace, 2 * HW, flowP, fres, 2 * HW);
        convLaunch(stream, t2, ctxDepth, depthP, H, W, 1, 1, ACT_SIGMOID);
      }
      off = streamMark;
    }
    off = levelMark;

    Carry* t = P; P = Cur; Cur = t;  // results of this level now in *P
  }

  // ---------------- full-resolution outputs ----------------
  const int HF = 192, WF = 640, HWf = HF * WF;
  float* o = (float*)d_out;
  float syf = 95.0f / 191.0f, sxf = 319.0f / 639.0f;
  EW(k_resize_bilinear, 2 * HWf, P->ff, o,            2, 96, 320, HF, WF, syf, sxf);
  EW(k_resize_bilinear, 2 * HWf, P->fb, o + 2 * HWf,  2, 96, 320, HF, WF, syf, sxf);
  EW(k_resize_bilinear, HWf,     P->d1, o + 4 * HWf,  1, 96, 320, HF, WF, syf, sxf);
  EW(k_resize_bilinear, HWf,     P->d2, o + 5 * HWf,  1, 96, 320, HF, WF, syf, sxf);

  float* wb = alloc((size_t)2 * HWf);
  float* wf = alloc((size_t)2 * HWf);
  EW(k_warp, 2 * HWf, o + 2 * HWf, o,            wb, 2, HF, WF, 0); // warp fb by ff
  EW(k_warp, 2 * HWf, o,           o + 2 * HWf,  wf, 2, HF, WF, 0); // warp ff by fb
  EW(k_final, HWf, o, wb, wf, HWf);
}
